// MCloss_25211458028169
// MI455X (gfx1250) — compile-verified, matrix operand order checked by reading, not run on hardware
//
#include <hip/hip_runtime.h>

#define N_BATCH 128
#define C_CLS   1024
#define NC      (N_BATCH * C_CLS)
#define EPS_F   1e-7f
#define LOGCLMP -100.0f

typedef __attribute__((ext_vector_type(16))) _Float16 v16h;
typedef __attribute__((ext_vector_type(8)))  float    v8f;

// 16-byte int vector type matching the async-load builtin's pointee type
// (diagnosed signature: int __vector_size__(4*sizeof(int)) AS1* src).
typedef int v4i_vs __attribute__((vector_size(4 * sizeof(int))));
typedef __attribute__((address_space(1))) v4i_vs* gptr_b128;
typedef __attribute__((address_space(3))) v4i_vs* lptr_b128;

// gfx1250 async global->LDS path (ASYNCcnt), with sync fallback if the
// toolchain doesn't declare the builtins.
#if defined(__gfx1250__) && \
    __has_builtin(__builtin_amdgcn_global_load_async_to_lds_b128) && \
    __has_builtin(__builtin_amdgcn_s_wait_asynccnt)
#define USE_ASYNC_LDS 1
#else
#define USE_ASYNC_LDS 0
#endif

// Issue one 64x64 R tile (row stride 68 floats in LDS) as 16B async copies.
__device__ __forceinline__ void load_R_tile(const float* __restrict__ Rg,
                                            float* __restrict__ dstbuf,
                                            int iBase, int jc, int tid) {
    #pragma unroll
    for (int l = tid; l < 64 * 16; l += 256) {
        const int row = l >> 4;
        const int col = (l & 15) << 2;
        const float* src = Rg + (size_t)(iBase + row) * C_CLS + jc + col;
        float* dst = dstbuf + row * 68 + col;
#if USE_ASYNC_LDS
        __builtin_amdgcn_global_load_async_to_lds_b128(
            (gptr_b128)(uintptr_t)src,
            (lptr_b128)(uintptr_t)dst,
            0, 0);
#else
        const float4 v = *(const float4*)src;
        dst[0] = v.x; dst[1] = v.y; dst[2] = v.z; dst[3] = v.w;
#endif
    }
}

// ---------------------------------------------------------------------------
// Kernel 1: MCM constraint (masked max over ancestors) + h_star.
//   mcm[n,i]     = max_j R[i,j] * sigmoid(h[n,j])
//   mcm_pos[n,i] = max_j R[i,j] * sigmoid(h[n,j]) * y[n,j]
//   h_star       = y ? mcm_pos : mcm
// 64 classes x 16 batch per block; R tile double-buffered via async DMA so the
// j-chunk DMA overlaps the mul+max compute of the previous chunk.
// ---------------------------------------------------------------------------
__global__ __launch_bounds__(256) void mcm_kernel(
    const float* __restrict__ Rg,
    const float* __restrict__ hg,
    const int*   __restrict__ yg,
    float*       __restrict__ mcm_out,
    float*       __restrict__ hstar)
{
    __shared__ float Rt[2][64 * 68];  // ping-pong R tiles (16B-aligned rows)
    __shared__ float pt[16 * 64];     // sigmoid(h)
    __shared__ float qt[16 * 64];     // sigmoid(h) * y

    const int tid   = threadIdx.x;
    const int iBase = blockIdx.x * 64;
    const int nBase = blockIdx.y * 16;
    const int ti    = tid & 63;
    const int grp   = tid >> 6;

    float m[4] = {0.f, 0.f, 0.f, 0.f};   // max over j includes R==0 zeros
    float q[4] = {0.f, 0.f, 0.f, 0.f};

    load_R_tile(Rg, Rt[0], iBase, 0, tid);   // prologue DMA
    int buf = 0;

    for (int jc = 0; jc < C_CLS; jc += 64) {
        // h/y 16x64 tile: fuse sigmoid + label mask while the R DMA flies
        for (int l = tid; l < 16 * 16; l += 256) {
            const int row = l >> 4;
            const int col = (l & 15) << 2;
            const size_t off = (size_t)(nBase + row) * C_CLS + jc + col;
            const float4 vh = *(const float4*)(hg + off);
            const int4   vy = *(const int4*)(yg + off);
            if (jc + 64 < C_CLS)   // next h tile toward L2 (global_prefetch_b8)
                __builtin_prefetch(hg + off + 64, 0, 0);
            float* dp = &pt[row * 64 + col];
            float* dq = &qt[row * 64 + col];
            const float p0 = 1.0f / (1.0f + __expf(-vh.x));
            const float p1 = 1.0f / (1.0f + __expf(-vh.y));
            const float p2 = 1.0f / (1.0f + __expf(-vh.z));
            const float p3 = 1.0f / (1.0f + __expf(-vh.w));
            dp[0] = p0; dp[1] = p1; dp[2] = p2; dp[3] = p3;
            dq[0] = vy.x ? p0 : 0.f;
            dq[1] = vy.y ? p1 : 0.f;
            dq[2] = vy.z ? p2 : 0.f;
            dq[3] = vy.w ? p3 : 0.f;
        }
#if USE_ASYNC_LDS
        __builtin_amdgcn_s_wait_asynccnt(0);   // this wave's R tile landed
#endif
        __syncthreads();                       // whole tile visible to block

        if (jc + 64 < C_CLS)                   // overlap next DMA with compute
            load_R_tile(Rg, Rt[buf ^ 1], iBase, jc + 64, tid);

        const float* __restrict__ Rcur = Rt[buf];
        #pragma unroll 4
        for (int jj = 0; jj < 64; jj += 4) {
            const float4 rv = *(const float4*)&Rcur[ti * 68 + jj];
            #pragma unroll
            for (int s = 0; s < 4; ++s) {
                const int nl = grp * 4 + s;
                const float4 pv = *(const float4*)&pt[nl * 64 + jj];
                const float4 qv = *(const float4*)&qt[nl * 64 + jj];
                m[s] = fmaxf(m[s], rv.x * pv.x);
                m[s] = fmaxf(m[s], rv.y * pv.y);
                m[s] = fmaxf(m[s], rv.z * pv.z);
                m[s] = fmaxf(m[s], rv.w * pv.w);
                q[s] = fmaxf(q[s], rv.x * qv.x);
                q[s] = fmaxf(q[s], rv.y * qv.y);
                q[s] = fmaxf(q[s], rv.z * qv.z);
                q[s] = fmaxf(q[s], rv.w * qv.w);
            }
        }
        __syncthreads();   // pt/qt (and fallback Rt) reuse next iteration
        buf ^= 1;
    }

    const int i = iBase + ti;
    #pragma unroll
    for (int s = 0; s < 4; ++s) {
        const int n = nBase + grp * 4 + s;
        const size_t idx = (size_t)n * C_CLS + i;
        mcm_out[idx] = m[s];
        hstar[idx]   = (yg[idx] != 0) ? q[s] : m[s];
    }
}

// ---------------------------------------------------------------------------
// Kernel 2: per-class dice terms + BCE, one wave per 16 classes.
// Batch column-sums via V_WMMA_F32_16X16X32_F16 with A == ones (f32 accum):
//   D[m, nCol] = sum_k B[k, nCol]
// B lane layout used here (contiguous k, valid as A == ones): N = lane%16, K = (lane<16 ? 0 : 16) + elem.
// ---------------------------------------------------------------------------
__global__ __launch_bounds__(32) void reduce_kernel(
    const float* __restrict__ hstar,
    const int*   __restrict__ yg,
    float*       __restrict__ acc)    // acc[0]=dice sum, acc[1]=bce inner sum
{
    const int lane  = threadIdx.x;
    const int cls   = blockIdx.x * 16 + (lane & 15);
    const int kHalf = (lane < 16) ? 0 : 16;

    v16h ones;
    #pragma unroll
    for (int e = 0; e < 16; ++e) ones[e] = (_Float16)1.0f;

    v8f c_inter = {};
    v8f c_card  = {};
    float bce = 0.0f;
    float ys  = 0.0f;

    #pragma unroll
    for (int nc = 0; nc < N_BATCH; nc += 32) {
        v16h bi, bc;
        #pragma unroll
        for (int e = 0; e < 16; ++e) {
            const int n = nc + kHalf + e;
            const size_t idx = (size_t)n * C_CLS + cls;
            const float hs = hstar[idx];
            const float yv = (yg[idx] != 0) ? 1.0f : 0.0f;
            bi[e] = (_Float16)(hs * yv);          // intersection term
            bc[e] = (_Float16)(hs + yv);          // cardinality term
            ys += yv;
            // hs in (0,1): log1p(-hs) ~= log(1-hs); both clamped at -100,
            // so use the hardware v_log_f32 path for both terms.
            const float lp  = fmaxf(__logf(hs),        LOGCLMP);
            const float l1p = fmaxf(__logf(1.0f - hs), LOGCLMP);
            bce += yv * lp + (1.0f - yv) * l1p;
        }
        c_inter = __builtin_amdgcn_wmma_f32_16x16x32_f16(
            false, ones, false, bi, (short)0, c_inter, false, false);
        c_card = __builtin_amdgcn_wmma_f32_16x16x32_f16(
            false, ones, false, bc, (short)0, c_card, false, false);
    }

    // class "present": lane L holds half the batch, lane L^16 the other half
    const float ys_tot = ys + __shfl_xor(ys, 16, 32);

    #pragma unroll
    for (int off = 16; off > 0; off >>= 1)
        bce += __shfl_down(bce, off, 32);
    if (lane == 0) atomicAdd(&acc[1], bce);

    if (lane < 16) {                  // D row M=0, N=lane -> column sums
        const float inter = c_inter[0];
        const float card  = c_card[0];
        const float dice  = 2.0f * inter / fmaxf(card, EPS_F);
        const float contrib = (ys_tot > 0.0f) ? (1.0f - dice) : 0.0f;
        atomicAdd(&acc[0], contrib);
    }
}

__global__ void init_kernel(float* __restrict__ acc) {
    if (threadIdx.x < 2) acc[threadIdx.x] = 0.0f;
}

__global__ void finalize_kernel(const float* __restrict__ acc,
                                float* __restrict__ out) {
    if (threadIdx.x == 0) {
        const float dice_loss = acc[0] / (float)C_CLS;
        const float bce       = -acc[1] / (float)NC;
        out[0] = dice_loss + bce;
    }
}

// ---------------------------------------------------------------------------
extern "C" void kernel_launch(void* const* d_in, const int* in_sizes, int n_in,
                              void* d_out, int out_size, void* d_ws, size_t ws_size,
                              hipStream_t stream) {
    (void)in_sizes; (void)n_in; (void)out_size; (void)ws_size;
    const float* h = (const float*)d_in[0];   // (128, 1024) f32 logits
    const int*   y = (const int*)d_in[1];     // (128, 1024) i32 labels
    const float* R = (const float*)d_in[2];   // (1024, 1024) f32 ancestry

    float* out   = (float*)d_out;             // [0]=loss, [1..]=mcm (N,C)
    float* hstar = (float*)d_ws;              // NC floats
    float* acc   = hstar + NC;                // 2 floats

    init_kernel<<<1, 32, 0, stream>>>(acc);

    dim3 g1(C_CLS / 64, N_BATCH / 16);        // (16, 8) blocks of 256
    mcm_kernel<<<g1, 256, 0, stream>>>(R, h, y, out + 1, hstar);

    reduce_kernel<<<C_CLS / 16, 32, 0, stream>>>(hstar, y, acc);

    finalize_kernel<<<1, 32, 0, stream>>>(acc, out);
}
